// QuantizedLinear_90975997263992
// MI455X (gfx1250) — matrix operand order repaired and checked by reading, not compiled, not run on hardware
//
#include <hip/hip_runtime.h>
#include <stdint.h>

// Quantized linear: y = (x @ (W_i8 * scale[n])) + bias[n]
// int8 weights are exact in f16 -> defer dequant past the matmul:
// f16 WMMA with fp32 accum, then *scale + bias in the epilogue.
#define IN_F   4096
#define OUT_F  16384
#define TOKENS 8192

#define BM 128      // output rows per block
#define BN 128      // output cols per block
#define KB 32       // K-depth per LDS chunk (= one WMMA k-step)
#define LDA 40      // padded f16 row stride (80B) -> breaks 64-bank pattern
#define NK (IN_F / KB)

typedef _Float16 v16h __attribute__((ext_vector_type(16)));
typedef _Float16 v8h  __attribute__((ext_vector_type(8)));
typedef _Float16 v4h  __attribute__((ext_vector_type(4)));
typedef _Float16 v2h  __attribute__((ext_vector_type(2)));
typedef float    v8f  __attribute__((ext_vector_type(8)));

union U32H2 { uint32_t u; v2h h; };

static __device__ __forceinline__ v16h cat16(v8h lo, v8h hi) {
  v16h r;
#pragma unroll
  for (int e = 0; e < 8; ++e) { r[e] = lo[e]; r[e + 8] = hi[e]; }
  return r;
}

struct Stage {
  float4 xa[4];   // 128x32 fp32 x-tile, 16 floats/thread
  uint2  wb[2];   // w bytes: rows k0 and k0+1, 8 cols each
};

__global__ __launch_bounds__(256) void qlinear_wmma_kernel(
    const float* __restrict__ x, const int8_t* __restrict__ w,
    const float* __restrict__ scales, const float* __restrict__ bias,
    float* __restrict__ out) {
  // double-buffered f16 staging tiles (A row-major, B column-major)
  __shared__ __align__(16) _Float16 As[2][BM * LDA];
  __shared__ __align__(16) _Float16 Bs[2][BN * LDA];

  const int tid   = threadIdx.x;
  const int rBase = blockIdx.y * BM;   // token rows
  const int nBase = blockIdx.x * BN;   // output cols

  // global->LDS staging mapping
  const int ar  = tid >> 3;            // x row within tile (0..31, x4 passes)
  const int ac4 = (tid & 7) * 4;       // x col (float4)
  const int bk2 = (tid >> 4) * 2;      // w k-pair base (0,2,..,30)
  const int bn8 = (tid & 15) * 8;      // w col start (8 int8)

  // wave tiling: 2 (M) x 4 (N) waves, each owns 64x32 outputs = 4x2 WMMA tiles
  const int lane = tid & 31;
  const int wv = tid >> 5;
  const int wm = wv >> 2;              // 0..1
  const int wn = wv & 3;               // 0..3
  const int hf = lane >> 4;            // half-wave select
  const int lm = lane & 15;

  auto load_stage = [&](int kc, Stage& s) {
    const float* xp = x + (long)(rBase + ar) * IN_F + kc * KB + ac4;
#pragma unroll
    for (int p = 0; p < 4; ++p)
      s.xa[p] = *(const float4*)(xp + (long)(32 * p) * IN_F);
    const uint8_t* wp =
        (const uint8_t*)w + (long)(kc * KB + bk2) * OUT_F + nBase + bn8;
    s.wb[0] = *(const uint2*)wp;
    s.wb[1] = *(const uint2*)(wp + OUT_F);
  };

  auto commit_stage = [&](int buf, const Stage& s) {
#pragma unroll
    for (int p = 0; p < 4; ++p) {
      v4h h;
      h[0] = (_Float16)s.xa[p].x; h[1] = (_Float16)s.xa[p].y;
      h[2] = (_Float16)s.xa[p].z; h[3] = (_Float16)s.xa[p].w;
      *(v4h*)&As[buf][(ar + 32 * p) * LDA + ac4] = h;
    }
    // int8 -> f16 via exponent magic: u = s^0x80; (0x6400|u) == 1024+u exact;
    // subtract 1152 (0xE480 packed) to recover s. 24 VALU ops / 16 bytes.
    const uint32_t a0 = s.wb[0].x ^ 0x80808080u;
    const uint32_t a1 = s.wb[0].y ^ 0x80808080u;   // k0 row, n0..n0+7
    const uint32_t c0 = s.wb[1].x ^ 0x80808080u;
    const uint32_t c1 = s.wb[1].y ^ 0x80808080u;   // k0+1 row
    // interleave to (k0,k1) byte pairs: [a.bn, c.bn, a.bn+1, c.bn+1]
    uint32_t t[4];
    t[0] = __builtin_amdgcn_perm(a0, c0, 0x01050004u);  // n0,n1
    t[1] = __builtin_amdgcn_perm(a0, c0, 0x03070206u);  // n2,n3
    t[2] = __builtin_amdgcn_perm(a1, c1, 0x01050004u);  // n4,n5
    t[3] = __builtin_amdgcn_perm(a1, c1, 0x03070206u);  // n6,n7
    U32H2 off; off.u = 0xE480E480u;                     // packed f16 -1152
#pragma unroll
    for (int q = 0; q < 4; ++q) {
      U32H2 lo, hi;
      lo.u = __builtin_amdgcn_perm(t[q], 0x64646464u, 0x00050004u);
      hi.u = __builtin_amdgcn_perm(t[q], 0x64646464u, 0x00070006u);
      v2h rl = lo.h + off.h;   // f16 pair (k0, k0+1) for col n0+2q
      v2h rh = hi.h + off.h;   // f16 pair (k0, k0+1) for col n0+2q+1
      *(v2h*)&Bs[buf][(bn8 + 2 * q + 0) * LDA + bk2] = rl;
      *(v2h*)&Bs[buf][(bn8 + 2 * q + 1) * LDA + bk2] = rh;
    }
  };

  v8f acc[4][2];
#pragma unroll
  for (int i = 0; i < 4; ++i)
#pragma unroll
    for (int j = 0; j < 2; ++j) acc[i][j] = {};

  Stage st;
  load_stage(0, st);
  commit_stage(0, st);
  __syncthreads();

  for (int kc = 0; kc < NK; ++kc) {
    const int  cur  = kc & 1;
    const bool more = (kc + 1) < NK;       // wave-uniform
    if (more) load_stage(kc + 1, st);      // global loads in flight over compute
    if (kc + 2 < NK) {                     // gfx1250 global_prefetch_b8
      __builtin_prefetch(x + (long)(rBase + ar) * IN_F + (kc + 2) * KB + ac4, 0, 1);
      __builtin_prefetch((const uint8_t*)w + (long)((kc + 2) * KB + bk2) * OUT_F + nBase + bn8, 0, 1);
    }

    // A frag (16x32 f16): lane=M row; K = {hf*8+0..7} U {16+hf*8+0..7}
    v16h a[4], b[2];
#pragma unroll
    for (int i = 0; i < 4; ++i) {
      const _Float16* p = &As[cur][(64 * wm + 16 * i + lm) * LDA + hf * 8];
      a[i] = cat16(*(const v8h*)p, *(const v8h*)(p + 16));
    }
    // B frag (32x16 f16): lane=N col; K = {hf*8+0..7} U {16+hf*8+0..7}
#pragma unroll
    for (int j = 0; j < 2; ++j) {
      const _Float16* p = &Bs[cur][(32 * wn + 16 * j + lm) * LDA + hf * 8];
      b[j] = cat16(*(const v8h*)p, *(const v8h*)(p + 16));
    }
#pragma unroll
    for (int i = 0; i < 4; ++i)
#pragma unroll
      for (int j = 0; j < 2; ++j)
        acc[i][j] = __builtin_amdgcn_wmma_f32_16x16x32_f16(
            false, a[i], false, b[j], (short)0, acc[i][j], false, false);

    if (more) commit_stage(cur ^ 1, st);   // fill other buffer
    __syncthreads();
  }

  // epilogue: deferred dequant + bias; D layout row = v + 8*hf, col = lm
  float sc[2], bi[2];
  int   col[2];
#pragma unroll
  for (int j = 0; j < 2; ++j) {
    col[j] = nBase + 32 * wn + 16 * j + lm;
    sc[j]  = scales[col[j]];
    bi[j]  = bias[col[j]];
  }
#pragma unroll
  for (int i = 0; i < 4; ++i)
#pragma unroll
    for (int j = 0; j < 2; ++j)
#pragma unroll
      for (int v = 0; v < 8; ++v) {
        int row = rBase + 64 * wm + 16 * i + 8 * hf + v;
        out[(long)row * OUT_F + col[j]] = acc[i][j][v] * sc[j] + bi[j];
      }
}

extern "C" void kernel_launch(void* const* d_in, const int* in_sizes, int n_in,
                              void* d_out, int out_size, void* d_ws, size_t ws_size,
                              hipStream_t stream) {
  const float*  x  = (const float*)d_in[0];
  const int8_t* wq = (const int8_t*)d_in[1];   // reference dtype is int8
  const float*  sc = (const float*)d_in[2];
  const float*  bs = (const float*)d_in[3];
  float* out = (float*)d_out;

  dim3 grid(OUT_F / BN, TOKENS / BM);  // 128 x 64 blocks
  qlinear_wmma_kernel<<<grid, dim3(256), 0, stream>>>(x, wq, sc, bs, out);
}
